// SAMN_45509473468543
// MI455X (gfx1250) — matrix operand order repaired and checked by reading, not compiled, not run on hardware
//
#include <hip/hip_runtime.h>
#include <hip/hip_bf16.h>

typedef __attribute__((ext_vector_type(16))) __bf16 v16bf;
typedef __attribute__((ext_vector_type(8)))  float  v8f;

#define D_DIM 128
#define M_SZ  8
#define A_SZ  32
#define NW    4   // waves per block in edge kernel

// ---------------- wave helpers (wave32) ----------------
__device__ __forceinline__ float wave_sum(float v) {
#pragma unroll
  for (int m = 16; m >= 1; m >>= 1) v += __shfl_xor(v, m, 32);
  return v;
}
__device__ __forceinline__ float wave_max(float v) {
#pragma unroll
  for (int m = 16; m >= 1; m >>= 1) v = fmaxf(v, __shfl_xor(v, m, 32));
  return v;
}
__device__ __forceinline__ int lower_bound_dev(const int* __restrict__ a, int n, int key) {
  int lo = 0, hi = n;
  while (lo < hi) {
    int mid = (lo + hi) >> 1;
    if (a[mid] < key) lo = mid + 1; else hi = mid;
  }
  return lo;
}

// ============================================================
// Kernel 1 (fused): per-edge memory attention -> f, then WMMA
// projection  hidden = relu(T@W1^T + F@W3^T + bias),
// beta = hidden @ h_w^T.  One wave handles a tile of 16 edges.
// ============================================================
__global__ __launch_bounds__(NW * 32) void samn_edge_kernel(
    const int* __restrict__ uid, const int* __restrict__ seg_ids,
    const int* __restrict__ trusteeid,
    const float* __restrict__ user_embed,
    const float* __restrict__ key_p,  // [128,8]
    const float* __restrict__ memb,   // [8,128]
    const float* __restrict__ w1_w, const float* __restrict__ w1_b,
    const float* __restrict__ w3_w, const float* __restrict__ w3_b,
    const float* __restrict__ bbias,  // [32]
    const float* __restrict__ h_w,    // [32]
    float* __restrict__ f_out,        // [E,128]
    float* __restrict__ beta_out,     // [E]
    int E)
{
  __shared__ __bf16 sT[NW][16][D_DIM];  // trust rows (bf16 for WMMA A)
  __shared__ float  sF[NW][16][D_DIM];  // friend vectors f (fp32)

  const int lane = threadIdx.x & 31;
  const int wave = threadIdx.x >> 5;
  const long long tb = ((long long)blockIdx.x * NW + wave) * 16;
  const int d0 = lane * 4;

  // ---------------- phase 1: per-edge memory attention ----------------
  for (int i = 0; i < 16; ++i) {
    long long er = tb + i;
    long long e  = (er < (long long)E) ? er : (long long)E - 1;
    const int sid = seg_ids[e];
    const int tu  = uid[sid];
    const int ru  = trusteeid[e];

    const float4 t4 = *(const float4*)(user_embed + (size_t)tu * D_DIM + d0);
    const float4 r4 = *(const float4*)(user_embed + (size_t)ru * D_DIM + d0);
    float tarr[4] = { t4.x, t4.y, t4.z, t4.w };
    float rarr[4] = { r4.x, r4.y, r4.z, r4.w };

    float nt = t4.x*t4.x + t4.y*t4.y + t4.z*t4.z + t4.w*t4.w;
    float nr = r4.x*r4.x + r4.y*r4.y + r4.z*r4.z + r4.w*r4.w;
    nt = wave_sum(nt);
    nr = wave_sum(nr);
    const float inv_norm = rsqrtf(nt) * rsqrtf(nr);

    // logits = (t*r/norm) @ key_p   -> [8]
    float p[M_SZ];
#pragma unroll
    for (int m = 0; m < M_SZ; ++m) p[m] = 0.f;
#pragma unroll
    for (int j = 0; j < 4; ++j) {
      const float sj = tarr[j] * rarr[j] * inv_norm;
      const float* kp = key_p + (size_t)(d0 + j) * M_SZ;
#pragma unroll
      for (int m = 0; m < M_SZ; ++m) p[m] += sj * kp[m];
    }
#pragma unroll
    for (int msk = 16; msk >= 1; msk >>= 1) {
#pragma unroll
      for (int m = 0; m < M_SZ; ++m) p[m] += __shfl_xor(p[m], msk, 32);
    }
    // softmax over 8 (redundant in every lane)
    float mx = p[0];
#pragma unroll
    for (int m = 1; m < M_SZ; ++m) mx = fmaxf(mx, p[m]);
    float asum = 0.f;
#pragma unroll
    for (int m = 0; m < M_SZ; ++m) { p[m] = __expf(p[m] - mx); asum += p[m]; }
    const float ainv = 1.0f / asum;

    // f = trustee * (a @ mem)
    float fv[4];
#pragma unroll
    for (int j = 0; j < 4; ++j) {
      float acc = 0.f;
#pragma unroll
      for (int m = 0; m < M_SZ; ++m) acc += p[m] * memb[m * D_DIM + d0 + j];
      fv[j] = rarr[j] * acc * ainv;
    }
#pragma unroll
    for (int j = 0; j < 4; ++j) {
      sT[wave][i][d0 + j] = (__bf16)tarr[j];
      sF[wave][i][d0 + j] = fv[j];
    }
    if (er < (long long)E) {
      float4 fo; fo.x = fv[0]; fo.y = fv[1]; fo.z = fv[2]; fo.w = fv[3];
      *(float4*)(f_out + (size_t)er * D_DIM + d0) = fo;
    }
  }
  __syncthreads();

  // ---------------- phase 2: WMMA projection -> beta ----------------
  // A layout (16-bit, 16x32): lane = half*16 + row; elems i: K=kb+16h+i
  // (contiguous, matching B).
  // B layout (16-bit, 32x16): lane = half*16 + n; elems i: K=kb+16h+i.
  const int half = lane >> 4;
  const int lid  = lane & 15;

  v8f acc0 = {};  // attention units N = 0..15
  v8f acc1 = {};  // attention units N = 16..31
#pragma unroll
  for (int c = 0; c < 4; ++c) {
    const int kb = c * 32;
    v16bf aT, aF;
#pragma unroll
    for (int q = 0; q < 8; ++q) {
      aT[q]     = sT[wave][lid][kb + 16 * half + q];
      aT[q + 8] = sT[wave][lid][kb + 16 * half + 8 + q];
      aF[q]     = (__bf16)sF[wave][lid][kb + 16 * half + q];
      aF[q + 8] = (__bf16)sF[wave][lid][kb + 16 * half + 8 + q];
    }
    const int krow = kb + 16 * half;
    const float* w1r0 = w1_w + (size_t)lid        * D_DIM + krow;
    const float* w1r1 = w1_w + (size_t)(lid + 16) * D_DIM + krow;
    const float* w3r0 = w3_w + (size_t)lid        * D_DIM + krow;
    const float* w3r1 = w3_w + (size_t)(lid + 16) * D_DIM + krow;
    v16bf b10, b11, b30, b31;
#pragma unroll
    for (int kk = 0; kk < 16; ++kk) {
      b10[kk] = (__bf16)w1r0[kk];
      b11[kk] = (__bf16)w1r1[kk];
      b30[kk] = (__bf16)w3r0[kk];
      b31[kk] = (__bf16)w3r1[kk];
    }
    acc0 = __builtin_amdgcn_wmma_f32_16x16x32_bf16(false, aT, false, b10,
                                                   (short)0, acc0, false, false);
    acc0 = __builtin_amdgcn_wmma_f32_16x16x32_bf16(false, aF, false, b30,
                                                   (short)0, acc0, false, false);
    acc1 = __builtin_amdgcn_wmma_f32_16x16x32_bf16(false, aT, false, b11,
                                                   (short)0, acc1, false, false);
    acc1 = __builtin_amdgcn_wmma_f32_16x16x32_bf16(false, aF, false, b31,
                                                   (short)0, acc1, false, false);
  }

  // C/D layout: lane=half*16+N(lid); elem v -> M = v + 8*half (edge in tile).
  const float bias0 = w1_b[lid]      + w3_b[lid]      + bbias[lid];
  const float bias1 = w1_b[lid + 16] + w3_b[lid + 16] + bbias[lid + 16];
  const float h0 = h_w[lid], h1 = h_w[lid + 16];
  float bp[8];
#pragma unroll
  for (int v = 0; v < 8; ++v)
    bp[v] = h0 * fmaxf(acc0[v] + bias0, 0.f) + h1 * fmaxf(acc1[v] + bias1, 0.f);
  // reduce over the 16 lanes of each half (sums over all 32 attention units)
#pragma unroll
  for (int msk = 8; msk >= 1; msk >>= 1) {
#pragma unroll
    for (int v = 0; v < 8; ++v) bp[v] += __shfl_xor(bp[v], msk, 32);
  }
  if (lid == 0) {
    const long long base = tb + 8 * half;
#pragma unroll
    for (int v = 0; v < 8; ++v)
      if (base + v < (long long)E) beta_out[base + v] = bp[v];
  }
}

// ============================================================
// Kernel 2: per-user segment softmax + weighted sum of f.
// One wave per user; seg_ids is sorted so ranges come from
// binary search (no atomics needed).
// ============================================================
__global__ __launch_bounds__(256) void samn_segment_kernel(
    const int* __restrict__ seg_ids, const float* __restrict__ beta,
    const float* __restrict__ f, float* __restrict__ ff, int B, int E)
{
  const int u = blockIdx.x * (blockDim.x >> 5) + (threadIdx.x >> 5);
  if (u >= B) return;
  const int lane = threadIdx.x & 31;

  const int lo = lower_bound_dev(seg_ids, E, u);
  const int hi = lower_bound_dev(seg_ids, E, u + 1);

  float mx = -__builtin_inff();
  for (int e = lo + lane; e < hi; e += 32) mx = fmaxf(mx, beta[e]);
  mx = wave_max(mx);

  float dsum = 0.f;
  for (int e = lo + lane; e < hi; e += 32) dsum += __expf(beta[e] - mx);
  dsum = wave_sum(dsum);
  const float dinv = (dsum == 0.f) ? 1.f : (1.f / dsum);

  const int d0 = lane * 4;
  float acc[4] = {0.f, 0.f, 0.f, 0.f};
  for (int e = lo; e < hi; ++e) {
    const float w = __expf(beta[e] - mx) * dinv;  // scalar broadcast load
    const float4 fe = *(const float4*)(f + (size_t)e * D_DIM + d0);
    acc[0] += w * fe.x; acc[1] += w * fe.y; acc[2] += w * fe.z; acc[3] += w * fe.w;
  }
  float4 o; o.x = acc[0]; o.y = acc[1]; o.z = acc[2]; o.w = acc[3];
  *(float4*)(ff + (size_t)u * D_DIM + d0) = o;
}

// ============================================================
// Kernel 3: BPR predictions. One wave per interaction.
// ============================================================
__global__ __launch_bounds__(256) void samn_pred_kernel(
    const int* __restrict__ uid, const int* __restrict__ user_idx,
    const int* __restrict__ ipos, const int* __restrict__ ineg,
    const float* __restrict__ user_embed, const float* __restrict__ item_embed,
    const float* __restrict__ ff, float* __restrict__ out, int N)
{
  const int n = blockIdx.x * (blockDim.x >> 5) + (threadIdx.x >> 5);
  if (n >= N) return;
  const int lane = threadIdx.x & 31;
  const int d0 = lane * 4;

  const int bidx = user_idx[n];
  const int tu   = uid[bidx];
  const float4 fv = *(const float4*)(ff + (size_t)bidx * D_DIM + d0);
  const float4 uv = *(const float4*)(user_embed + (size_t)tu * D_DIM + d0);
  const float ux = fv.x + uv.x, uy = fv.y + uv.y, uz = fv.z + uv.z, uw = fv.w + uv.w;

  const float4 pv = *(const float4*)(item_embed + (size_t)ipos[n] * D_DIM + d0);
  const float4 nv = *(const float4*)(item_embed + (size_t)ineg[n] * D_DIM + d0);
  float pp = ux * pv.x + uy * pv.y + uz * pv.z + uw * pv.w;
  float pn = ux * nv.x + uy * nv.y + uz * nv.z + uw * nv.w;
  pp = wave_sum(pp);
  pn = wave_sum(pn);
  if (lane == 0) {
    out[n] = pp;
    out[(size_t)N + n] = pn;
  }
}

// ============================================================
extern "C" void kernel_launch(void* const* d_in, const int* in_sizes, int n_in,
                              void* d_out, int out_size, void* d_ws, size_t ws_size,
                              hipStream_t stream) {
  const int*   uid        = (const int*)d_in[0];
  const int*   seg_ids    = (const int*)d_in[1];
  const int*   trusteeid  = (const int*)d_in[2];
  const int*   user_idx   = (const int*)d_in[3];
  const int*   ipos       = (const int*)d_in[4];
  const int*   ineg       = (const int*)d_in[5];
  const float* user_embed = (const float*)d_in[6];
  const float* item_embed = (const float*)d_in[7];
  const float* key_p      = (const float*)d_in[8];
  const float* memb       = (const float*)d_in[9];
  const float* w1_w       = (const float*)d_in[10];
  const float* w1_b       = (const float*)d_in[11];
  const float* w3_w       = (const float*)d_in[12];
  const float* w3_b       = (const float*)d_in[13];
  const float* bbias      = (const float*)d_in[14];
  const float* h_w        = (const float*)d_in[15];

  const int B = in_sizes[0];
  const int E = in_sizes[1];
  const int N = in_sizes[3];

  // workspace layout: f [E,128] fp32 | beta [E] | ff [B,128]
  float* ws_f = (float*)d_ws;
  float* beta = (float*)((char*)d_ws + (size_t)E * D_DIM * sizeof(float));
  float* ff   = beta + E;
  float* out  = (float*)d_out;

  const int tiles   = (E + 15) / 16;
  const int blocks1 = (tiles + NW - 1) / NW;
  samn_edge_kernel<<<blocks1, NW * 32, 0, stream>>>(
      uid, seg_ids, trusteeid, user_embed, key_p, memb,
      w1_w, w1_b, w3_w, w3_b, bbias, h_w, ws_f, beta, E);

  const int wpb = 8;  // waves per block
  samn_segment_kernel<<<(B + wpb - 1) / wpb, wpb * 32, 0, stream>>>(
      seg_ids, beta, ws_f, ff, B, E);

  samn_pred_kernel<<<(N + wpb - 1) / wpb, wpb * 32, 0, stream>>>(
      uid, user_idx, ipos, ineg, user_embed, item_embed, ff, out, N);
}
